// GAT_63969242906975
// MI455X (gfx1250) — matrix operand order repaired and checked by reading, not compiled, not run on hardware
//
#include <hip/hip_runtime.h>
#include <cstdint>

#ifndef __has_builtin
#define __has_builtin(x) 0
#endif

namespace {
constexpr int Bc  = 4;
constexpr int Nn  = 10000;
constexpr int Ee  = 100000;
constexpr int Hh  = 8;
constexpr int Ff  = 32;
constexpr int FIN = 256;
constexpr int LDW = 264;           // padded f16 row stride for W (in halves): 528B, 16B aligned
constexpr int BN  = Bc * Nn;       // 40000 rows of x
constexpr int BEr = Bc * Ee;       // 400000 rows of rel
}

typedef __attribute__((ext_vector_type(16))) _Float16 v16h;
typedef __attribute__((ext_vector_type(8)))  _Float16 v8h;
typedef __attribute__((ext_vector_type(8)))  float    v8f;
typedef __attribute__((ext_vector_type(4)))  float    f4;
typedef __attribute__((ext_vector_type(4)))  unsigned tdm_u32x4;
typedef __attribute__((ext_vector_type(4)))  int      tdm_i32x4;
typedef __attribute__((ext_vector_type(8)))  int      tdm_i32x8;

__device__ __forceinline__ v8f wmma_f16(v16h a, v16h b, v8f c) {
  // (neg_a, A, neg_b, B, c_mod, C, reuse_a, reuse_b)
  return __builtin_amdgcn_wmma_f32_16x16x32_f16(false, a, false, b, (short)0, c, false, false);
}

// A-fragment (16-bit A 16x32 layout): lane half hi, af[j] = x[k0 + 8*hi + j] (j<8),
// af[8+j] = x[k0 + 16 + 8*hi + j]
__device__ __forceinline__ v16h cvtA(f4 a0, f4 a1, f4 a2, f4 a3) {
  v16h r;
  r[0]=(_Float16)a0[0];  r[1]=(_Float16)a0[1];  r[2]=(_Float16)a0[2];  r[3]=(_Float16)a0[3];
  r[4]=(_Float16)a1[0];  r[5]=(_Float16)a1[1];  r[6]=(_Float16)a1[2];  r[7]=(_Float16)a1[3];
  r[8]=(_Float16)a2[0];  r[9]=(_Float16)a2[1];  r[10]=(_Float16)a2[2]; r[11]=(_Float16)a2[3];
  r[12]=(_Float16)a3[0]; r[13]=(_Float16)a3[1]; r[14]=(_Float16)a3[2]; r[15]=(_Float16)a3[3];
  return r;
}

// B-fragment (B 32x16 layout, matches A): lane n, half hi:
// bf[j] = W[n][k0 + 8*hi + j] (j<8), bf[8+j] = W[n][k0 + 16 + 8*hi + j]
__device__ __forceinline__ v16h loadB(const _Float16* Wl, int nloc, int k0, int hi) {
  const v8h* p = (const v8h*)(Wl + (size_t)nloc * LDW + k0 + 8 * hi);
  v8h x0 = p[0], x1 = p[2];
  return __builtin_shufflevector(x0, x1, 0,1,2,3,4,5,6,7,8,9,10,11,12,13,14,15);
}

// Stage one 64-column slice of W (64*LDW f16 = 33792 B, contiguous in global) into LDS.
// Uses the Tensor Data Mover (one descriptor, wave 0 issues; TDM ignores EXEC), with a
// plain vector-copy fallback for toolchains without the builtin (and for the host pass).
__device__ __forceinline__ void stage_w_lds(_Float16* Wl, const _Float16* gsrc) {
#if defined(__gfx1250__) && __has_builtin(__builtin_amdgcn_tensor_load_to_lds)
  if (threadIdx.x == 0) {
    unsigned lds_addr = (unsigned)(uintptr_t)(void*)Wl;        // flat low 32 = LDS byte addr
    unsigned long long ga = (unsigned long long)(uintptr_t)gsrc;
    constexpr int n8 = 64 * LDW * 2 / 8;                        // 4224 x 8-byte elements
    tdm_u32x4 g0;
    g0[0] = 1u;                                                 // count=1 (valid user D#)
    g0[1] = lds_addr;                                           // D#.lds_addr
    g0[2] = (unsigned)(ga & 0xffffffffull);                     // global_addr[31:0]
    g0[3] = (unsigned)((ga >> 32) & 0x01ffffffull) | (2u << 30);// global_addr[56:32]|type=2
    tdm_i32x8 g1;
    g1[0] = 3 << 16;                                            // data_size=8B, no mask/pad
    g1[1] = (n8 & 0xffff) << 16;                                // tensor_dim0[15:0]
    g1[2] = ((n8 >> 16) & 0xffff) | (1 << 16);                  // tensor_dim0[31:16]|dim1=1
    g1[3] = (n8 & 0xffff) << 16;                                // tile_dim0 = n8
    g1[4] = 0;                                                  // tile_dim1/2 unused
    g1[5] = n8;                                                 // tensor_dim0_stride[31:0]
    g1[6] = 0;
    g1[7] = 0;
    tdm_i32x4 z4 = (tdm_i32x4)0;
#if __has_include(<hip/amd_detail/amd_gfx1250_TDM.h>)
    tdm_i32x8 z8 = (tdm_i32x8)0;
    __builtin_amdgcn_tensor_load_to_lds(g0, g1, z4, z4, z8, 0);
#else
    __builtin_amdgcn_tensor_load_to_lds(g0, g1, z4, z4, 0);
#endif
  }
#if __has_builtin(__builtin_amdgcn_s_wait_tensorcnt)
  __builtin_amdgcn_s_wait_tensorcnt(0);                         // NOP on non-issuing waves
#endif
#else
  const uint4* g = (const uint4*)gsrc;
  uint4* l = (uint4*)Wl;
  for (int i = threadIdx.x; i < 64 * LDW / 8; i += 256) l[i] = g[i];
#endif
  __syncthreads();
}

__device__ __forceinline__ unsigned encf(float f) {
  unsigned u = __float_as_uint(f);
  return (u & 0x80000000u) ? ~u : (u | 0x80000000u);
}
__device__ __forceinline__ float decf(unsigned e) {
  unsigned u = (e & 0x80000000u) ? (e & 0x7fffffffu) : ~e;
  return __uint_as_float(u);
}

// ---------------- W -> f16 with padded stride --------------------------------
__global__ void __launch_bounds__(256) k_convert_w(const float* __restrict__ W,
                                                   _Float16* __restrict__ Whf) {
  int i = blockIdx.x * 256 + threadIdx.x;      // 256*264 = 67584 elements
  if (i >= 256 * LDW) return;
  int r = i / LDW, c = i - r * LDW;
  Whf[i] = (c < FIN) ? (_Float16)W[r * FIN + c] : (_Float16)0.f;
}

// ---------------- init: acc = x, denom = 0, maxb = enc(smallest) -------------
__global__ void __launch_bounds__(256) k_init(const float* __restrict__ x,
                                              float* __restrict__ accb,
                                              float* __restrict__ denom,
                                              unsigned* __restrict__ maxb) {
  int i = blockIdx.x * 256 + threadIdx.x;
  if (i < BN * FIN) accb[i] = x[i];
  if (i < BN * Hh)  denom[i] = 0.f;
  if (i < Bc)       maxb[i] = 0u;
}

// ---------------- GEMM: out[M,256] = A[M,256] @ W^T (f16 WMMA, f32 acc) ------
__global__ void __launch_bounds__(256) k_gemm(const float* __restrict__ A,
                                              const _Float16* __restrict__ Whf,
                                              float* __restrict__ out, int M) {
  __shared__ _Float16 Wl[64 * LDW];            // 33792 B: 64 output columns of W
  const int c0 = blockIdx.y * 64;
  stage_w_lds(Wl, Whf + (size_t)c0 * LDW);

  const int wave = threadIdx.x >> 5, lane = threadIdx.x & 31;
  const int m = lane & 15, hi = lane >> 4;
  const int row0 = blockIdx.x * 128 + wave * 16;
  int ra = row0 + m; if (ra > M - 1) ra = M - 1;
  const float* arow = A + (size_t)ra * FIN;

  v8f acc[4];
#pragma unroll
  for (int t = 0; t < 4; ++t) acc[t] = (v8f)0.0f;

  for (int k0 = 0; k0 < FIN; k0 += 32) {
    if (k0 + 32 < FIN) __builtin_prefetch(arow + k0 + 32 + 8 * hi, 0, 3);
    const float* ap = arow + k0 + 8 * hi;
    f4 a0 = *(const f4*)(ap);
    f4 a1 = *(const f4*)(ap + 4);
    f4 a2 = *(const f4*)(ap + 16);
    f4 a3 = *(const f4*)(ap + 20);
    v16h af = cvtA(a0, a1, a2, a3);
#pragma unroll
    for (int t = 0; t < 4; ++t) {
      v16h bf = loadB(Wl, t * 16 + m, k0, hi);
      acc[t] = wmma_f16(af, bf, acc[t]);
    }
  }

#pragma unroll
  for (int t = 0; t < 4; ++t)
#pragma unroll
    for (int v = 0; v < 8; ++v) {
      int r = row0 + v + 8 * hi;
      if (r < M) out[(size_t)r * FIN + c0 + t * 16 + m] = acc[t][v];
    }
}

// ---------------- per-node attention scores ----------------------------------
__global__ void __launch_bounds__(256) k_scores_node(const float* __restrict__ proj,
                                                     const float* __restrict__ a_src,
                                                     const float* __restrict__ a_trg,
                                                     float* __restrict__ ssrc,
                                                     float* __restrict__ strg) {
  int i = blockIdx.x * 256 + threadIdx.x;      // B*N*H = 320000
  if (i >= BN * Hh) return;
  int h = i & (Hh - 1);
  const float* p = proj + (size_t)(i >> 3) * FIN + h * Ff;
  float ss = 0.f, st = 0.f;
#pragma unroll
  for (int f = 0; f < Ff; ++f) {
    float v = p[f];
    ss += v * a_src[h * Ff + f];
    st += v * a_trg[h * Ff + f];
  }
  ssrc[i] = ss;
  strg[i] = st;
}

// ---------------- edge scores + leaky relu + per-batch max -------------------
__global__ void __launch_bounds__(256) k_edge_scores(const int* __restrict__ ei,
                                                     const float* __restrict__ ssrc,
                                                     const float* __restrict__ strg,
                                                     float* __restrict__ se,
                                                     unsigned* __restrict__ maxb) {
  int i = blockIdx.x * 256 + threadIdx.x;      // B*E*H = 3,200,000 (block spans one batch)
  int b = i / (Ee * Hh);
  int r = i - b * (Ee * Hh);
  int e = r >> 3, h = r & (Hh - 1);
  int src = ei[b * 2 * Ee + e];
  int trg = ei[b * 2 * Ee + Ee + e];
  float s = ssrc[(b * Nn + src) * Hh + h] + strg[(b * Nn + trg) * Hh + h];
  s = (s > 0.f) ? s : 0.2f * s;                // leaky_relu(0.2)
  se[i] = s;

  __shared__ float red[256];
  red[threadIdx.x] = s;
  __syncthreads();
  for (int off = 128; off > 0; off >>= 1) {
    if (threadIdx.x < off) red[threadIdx.x] = fmaxf(red[threadIdx.x], red[threadIdx.x + off]);
    __syncthreads();
  }
  if (threadIdx.x == 0) atomicMax(&maxb[b], encf(red[0]));
}

// ---------------- exp(s - max) + segment-sum of denominators -----------------
__global__ void __launch_bounds__(256) k_expden(const int* __restrict__ ei,
                                                const unsigned* __restrict__ maxb,
                                                float* __restrict__ se,
                                                float* __restrict__ denom) {
  int i = blockIdx.x * 256 + threadIdx.x;      // B*E*H
  int b = i / (Ee * Hh);
  int r = i - b * (Ee * Hh);
  int e = r >> 3, h = r & (Hh - 1);
  float ex = expf(se[i] - decf(maxb[b]));
  se[i] = ex;
  int trg = ei[b * 2 * Ee + Ee + e];
  atomicAdd(&denom[(b * Nn + trg) * Hh + h], ex);
}

// ---------------- fused: rel_proj GEMM + message build + scatter-add ---------
__global__ void __launch_bounds__(256) k_gemm_rel_scatter(const float* __restrict__ REL,
                                                          const _Float16* __restrict__ Whf,
                                                          const int* __restrict__ ei,
                                                          const float* __restrict__ proj,
                                                          const float* __restrict__ expw,
                                                          const float* __restrict__ denom,
                                                          float* __restrict__ accb) {
  __shared__ _Float16 Wl[64 * LDW];
  const int c0 = blockIdx.y * 64;
  stage_w_lds(Wl, Whf + (size_t)c0 * LDW);

  const int wave = threadIdx.x >> 5, lane = threadIdx.x & 31;
  const int m = lane & 15, hi = lane >> 4;
  const int row0 = blockIdx.x * 128 + wave * 16;       // BEr % 128 == 0: no tail
  const float* arow = REL + (size_t)(row0 + m) * FIN;

  v8f acc[4];
#pragma unroll
  for (int t = 0; t < 4; ++t) acc[t] = (v8f)0.0f;

  for (int k0 = 0; k0 < FIN; k0 += 32) {
    if (k0 + 32 < FIN) __builtin_prefetch(arow + k0 + 32 + 8 * hi, 0, 3);
    const float* ap = arow + k0 + 8 * hi;
    f4 a0 = *(const f4*)(ap);
    f4 a1 = *(const f4*)(ap + 4);
    f4 a2 = *(const f4*)(ap + 16);
    f4 a3 = *(const f4*)(ap + 20);
    v16h af = cvtA(a0, a1, a2, a3);
#pragma unroll
    for (int t = 0; t < 4; ++t) {
      v16h bf = loadB(Wl, t * 16 + m, k0, hi);
      acc[t] = wmma_f16(af, bf, acc[t]);
    }
  }

  // Epilogue: acc[t][v] = rel_proj[row0+v+8*hi][c0+t*16+m]
  // msg = proj[src]*att + rel_proj ; accb[trg] += msg   (16-lane halves share an edge
  // per v -> 64B-contiguous gather & atomics)
#pragma unroll
  for (int t = 0; t < 4; ++t) {
    int col = c0 + t * 16 + m;
    int head = col >> 5;                                 // F = 32
#pragma unroll
    for (int v = 0; v < 8; ++v) {
      int r = row0 + v + 8 * hi;                         // global edge row (b*E+e)
      int b = r / Ee;
      int e = r - b * Ee;
      int src = ei[b * 2 * Ee + e];
      int trg = ei[b * 2 * Ee + Ee + e];
      float den = denom[(b * Nn + trg) * Hh + head] + 1e-16f;
      float att = expw[(size_t)r * Hh + head] / den;
      float msg = acc[t][v] + att * proj[(size_t)(b * Nn + src) * FIN + col];
      atomicAdd(&accb[(size_t)(b * Nn + trg) * FIN + col], msg);
    }
  }
}

// ---------------- finalize: elu(acc + bias) ----------------------------------
__global__ void __launch_bounds__(256) k_final(const float* __restrict__ accb,
                                               const float* __restrict__ bias,
                                               float* __restrict__ out) {
  int i = blockIdx.x * 256 + threadIdx.x;
  if (i >= BN * FIN) return;
  float v = accb[i] + bias[i & (FIN - 1)];
  out[i] = (v > 0.f) ? v : (expf(v) - 1.f);
}

extern "C" void kernel_launch(void* const* d_in, const int* in_sizes, int n_in,
                              void* d_out, int out_size, void* d_ws, size_t ws_size,
                              hipStream_t stream) {
  const float* x     = (const float*)d_in[0];   // (B,N,FIN)
  const int*   ei    = (const int*)  d_in[1];   // (B,2,E)
  const float* rel   = (const float*)d_in[2];   // (B,E,FIN)
  const float* W     = (const float*)d_in[3];   // (FIN,FIN)
  const float* a_src = (const float*)d_in[4];   // (1,H,F)
  const float* a_trg = (const float*)d_in[5];   // (1,H,F)
  const float* bias  = (const float*)d_in[6];   // (FIN,)
  float* out = (float*)d_out;

  char* ws = (char*)d_ws;
  size_t off = 0;
  _Float16* Whf  = (_Float16*)(ws + off); off += (size_t)256 * LDW * 2;   // 135,168 B
  float* proj    = (float*)(ws + off);    off += (size_t)BN * FIN * 4;    // 40.96 MB
  float* accb    = (float*)(ws + off);    off += (size_t)BN * FIN * 4;    // 40.96 MB
  float* ssrc    = (float*)(ws + off);    off += (size_t)BN * Hh * 4;
  float* strg    = (float*)(ws + off);    off += (size_t)BN * Hh * 4;
  float* se      = (float*)(ws + off);    off += (size_t)BEr * Hh * 4;    // 12.8 MB
  float* denom   = (float*)(ws + off);    off += (size_t)BN * Hh * 4;
  unsigned* maxb = (unsigned*)(ws + off); off += 256;
  (void)in_sizes; (void)n_in; (void)out_size; (void)ws_size;

  k_convert_w<<<(256 * LDW) / 256, 256, 0, stream>>>(W, Whf);
  k_init<<<(BN * FIN + 255) / 256, 256, 0, stream>>>(x, accb, denom, maxb);
  k_gemm<<<dim3((BN + 127) / 128, 4), 256, 0, stream>>>(x, Whf, proj, BN);
  k_scores_node<<<(BN * Hh + 255) / 256, 256, 0, stream>>>(proj, a_src, a_trg, ssrc, strg);
  k_edge_scores<<<(BEr * Hh) / 256, 256, 0, stream>>>(ei, ssrc, strg, se, maxb);
  k_expden<<<(BEr * Hh) / 256, 256, 0, stream>>>(ei, maxb, se, denom);
  k_gemm_rel_scatter<<<dim3(BEr / 128, 4), 256, 0, stream>>>(rel, Whf, ei, proj, se, denom, accb);
  k_final<<<(BN * FIN + 255) / 256, 256, 0, stream>>>(accb, bias, out);
}
